// CausalSelfAttention_584115553069
// MI455X (gfx1250) — matrix operand order repaired and checked by reading, not compiled, not run on hardware
//
#include <hip/hip_runtime.h>
#include <hip/hip_bf16.h>

typedef __bf16 bf16_t;
typedef __attribute__((ext_vector_type(16))) __bf16 v16bf;
typedef __attribute__((ext_vector_type(8)))  __bf16 v8bf;
typedef __attribute__((ext_vector_type(8)))  float  v8f;
typedef int vint4 __attribute__((vector_size(16)));

#define BB 4
#define TT 2048
#define CC 1024
#define HH 16
#define DD 64
#define NEGBIG (-1.0e9f)

#define AS1 __attribute__((address_space(1)))
#define AS3 __attribute__((address_space(3)))

#if defined(__AMDGCN__) && __has_builtin(__builtin_amdgcn_global_load_async_to_lds_b128)
#define HAVE_ASYNC_LDS 1
#else
#define HAVE_ASYNC_LDS 0
#endif

// ---------- async copy helpers (16B global -> LDS) ----------

__device__ __forceinline__ void async_copy16(const bf16_t* g, bf16_t* l) {
#if HAVE_ASYNC_LDS
  __builtin_amdgcn_global_load_async_to_lds_b128(
      (AS1 vint4*)(unsigned long long)g,
      (AS3 vint4*)(unsigned int)(unsigned long long)l,
      0, 0);
#else
  *(v8bf*)l = *(const v8bf*)g;
#endif
}

template <int N>
__device__ __forceinline__ void async_wait() {
#if HAVE_ASYNC_LDS
#if __has_builtin(__builtin_amdgcn_s_wait_asynccnt)
  __builtin_amdgcn_s_wait_asynccnt(N);
#else
  asm volatile("s_wait_asynccnt %0" ::"i"(N) : "memory");
#endif
#endif
}

// ---------- WMMA helpers ----------

__device__ __forceinline__ v8f wmma_bf16(v16bf a, v16bf b, v8f c) {
  return __builtin_amdgcn_wmma_f32_16x16x32_bf16(
      /*neg_a=*/false, a, /*neg_b=*/false, b,
      /*c_mod=*/(short)0, c, /*reuse_a=*/false, /*reuse_b=*/false);
}

// A-matrix 16x32 bf16 fragment (ISA 7.12.2): lane m = lane&15, half = lane>>4.
// VGPR0..3 hold K = half*8..+7 ; VGPR4..7 hold K = 16+half*8..+7.
__device__ __forceinline__ v16bf load_a_frag(const bf16_t* p, int ld, int lane) {
  const int m  = lane & 15;
  const int kh = lane >> 4;
  const bf16_t* r = p + (size_t)m * ld + kh * 8;
  v8bf lo = *(const v8bf*)(r);
  v8bf hi = *(const v8bf*)(r + 16);
  return __builtin_shufflevector(lo, hi, 0,1,2,3,4,5,6,7,8,9,10,11,12,13,14,15);
}

// B-matrix 32x16 bf16 fragment, sourced from B^T stored row-major [N x K]:
// lane column n = lane&15, half = lane>>4; elems 0..7 hold K = half*8..+7, elems 8..15 hold K = 16+half*8..+7.
__device__ __forceinline__ v16bf load_b_frag(const bf16_t* p, int ld, int lane) {
  const int n  = lane & 15;
  const int kh = lane >> 4;
  const bf16_t* r = p + (size_t)n * ld + kh * 8;
  v8bf lo = *(const v8bf*)(r);
  v8bf hi = *(const v8bf*)(r + 16);
  return __builtin_shufflevector(lo, hi, 0,1,2,3,4,5,6,7,8,9,10,11,12,13,14,15);
}

// ---------- prep kernels ----------

__global__ void __launch_bounds__(256) k_to_bf16(const float* __restrict__ src,
                                                 bf16_t* __restrict__ dst, int n) {
  int i = blockIdx.x * 256 + threadIdx.x;
  if (i < n) dst[i] = (bf16_t)src[i];
}

// W [R x Cc] row-major -> WT [Cc x R] row-major bf16
__global__ void __launch_bounds__(256) k_transpose_bf16(const float* __restrict__ W,
                                                        bf16_t* __restrict__ WT,
                                                        int R, int Cc) {
  int i = blockIdx.x * 256 + threadIdx.x;
  if (i < R * Cc) {
    int r = i / Cc, c = i - r * Cc;
    WT[(size_t)c * R + r] = (bf16_t)W[i];
  }
}

// ---------- unified GEMM: C = A[ M x 1024 ] @ BT^T + bias ----------
// grid (N/128, M/128), 256 threads = 8 waves; wave tile 32x64.
// K staged in 64-wide slabs through LDS, double-buffered async copies.
// QKV=true  -> N=3072, epilogue scatters q (bf16) / k,v (f32 + bf16 copies)
// QKV=false -> N=1024, epilogue writes f32 out + bias
template <bool QKV>
__global__ void __launch_bounds__(256)
k_gemm(const bf16_t* __restrict__ A, const bf16_t* __restrict__ BT,
       const float* __restrict__ bias,
       float* __restrict__ out,
       bf16_t* __restrict__ qb, float* __restrict__ kout, bf16_t* __restrict__ kb,
       float* __restrict__ vout, bf16_t* __restrict__ vTb)
{
  __shared__ __align__(16) bf16_t As[2][128 * 64];   // 2 x 16KB
  __shared__ __align__(16) bf16_t Bs[2][128 * 64];   // 2 x 16KB

  const int tid  = threadIdx.x;
  const int lane = tid & 31, wave = tid >> 5;
  const int wr = wave >> 1, wc = wave & 1;
  const int mblk = blockIdx.y * 128, nblk = blockIdx.x * 128;
  const int nc = lane & 15, half = lane >> 4;

  // cooperative copy mapping: thread -> (row 0..127, 64B half-row)
  const int crow = tid >> 1;
  const int ccol = (tid & 1) * 32;  // elements

  auto stage_load = [&](int buf, int k0) {
    const bf16_t* ga = A + (size_t)(mblk + crow) * CC + k0 + ccol;
    const bf16_t* gb = BT + (size_t)(nblk + crow) * CC + k0 + ccol;
#pragma unroll
    for (int o = 0; o < 32; o += 8) {
      async_copy16(ga + o, &As[buf][crow * 64 + ccol + o]);
      async_copy16(gb + o, &Bs[buf][crow * 64 + ccol + o]);
    }
  };

  v8f acc[2][4];
#pragma unroll
  for (int ci = 0; ci < 4; ++ci) {
    float bv = bias[nblk + wc * 64 + ci * 16 + nc];
    v8f t = {bv, bv, bv, bv, bv, bv, bv, bv};
    acc[0][ci] = t;
    acc[1][ci] = t;
  }

  stage_load(0, 0);
  const int NK = CC / 64;  // 16 slabs
  for (int kt = 0; kt < NK; ++kt) {
    const int cur = kt & 1;
    if (kt + 1 < NK) {
      stage_load(cur ^ 1, (kt + 1) * 64);  // prefetch next slab
      async_wait<8>();                     // current slab (issued earlier) done
    } else {
      async_wait<0>();
    }
    __syncthreads();

#pragma unroll
    for (int ks = 0; ks < 2; ++ks) {
      v16bf a0 = load_a_frag(&As[cur][(wr * 32) * 64 + ks * 32], 64, lane);
      v16bf a1 = load_a_frag(&As[cur][(wr * 32 + 16) * 64 + ks * 32], 64, lane);
      v16bf bfr[4];
#pragma unroll
      for (int ci = 0; ci < 4; ++ci)
        bfr[ci] = load_b_frag(&Bs[cur][(wc * 64 + ci * 16) * 64 + ks * 32], 64, lane);
#pragma unroll
      for (int ci = 0; ci < 4; ++ci) {
        acc[0][ci] = wmma_bf16(a0, bfr[ci], acc[0][ci]);
        acc[1][ci] = wmma_bf16(a1, bfr[ci], acc[1][ci]);
      }
    }
    __syncthreads();  // slab fully consumed before it is overwritten
  }

  const int m0 = mblk + wr * 32;
  const int n0 = nblk + wc * 64;
  if (QKV) {
    const int sect = n0 >> 10;  // block fully inside q/k/v (128 | 1024)
#pragma unroll
    for (int ri = 0; ri < 2; ++ri) {
#pragma unroll
      for (int ci = 0; ci < 4; ++ci) {
#pragma unroll
        for (int r = 0; r < 8; ++r) {
          int m = m0 + ri * 16 + r + 8 * half;  // global row (b*T + t)
          int n = n0 + ci * 16 + nc;            // global col in [0,3072)
          float val = acc[ri][ci][r];
          int b = m >> 11, t = m & (TT - 1);
          int c = n & (CC - 1);
          int h = c >> 6, d = c & (DD - 1);
          size_t idx = ((size_t)(b * HH + h) * TT + t) * DD + d;
          if (sect == 0) {
            qb[idx] = (bf16_t)val;
          } else if (sect == 1) {
            kout[idx] = val;
            kb[idx] = (bf16_t)val;
          } else {
            vout[idx] = val;
            vTb[((size_t)(b * HH + h) * DD + d) * TT + t] = (bf16_t)val;
          }
        }
      }
    }
  } else {
#pragma unroll
    for (int ri = 0; ri < 2; ++ri)
#pragma unroll
      for (int ci = 0; ci < 4; ++ci)
#pragma unroll
        for (int r = 0; r < 8; ++r) {
          int m = m0 + ri * 16 + r + 8 * half;
          int n = n0 + ci * 16 + nc;
          out[(size_t)m * CC + n] = acc[ri][ci][r];
        }
  }
}

// ---------- flash attention ----------
// grid (T/64, B*H), 128 threads = 4 waves; each wave owns 16 q-rows.
// k-tile [64x64] and vT-tile [64x64] double-buffered in LDS via async copies.
__global__ void __launch_bounds__(128)
k_attn(const bf16_t* __restrict__ qb,   // [B,H,T,D] bf16
       const bf16_t* __restrict__ kb,   // [B,H,T,D] bf16
       const bf16_t* __restrict__ vTb,  // [B,H,D,T] bf16
       bf16_t* __restrict__ ob)         // attn out bf16 [B,T,C]
{
  __shared__ __align__(16) bf16_t Ks[2][64 * 64];    // [t_rel][d]
  __shared__ __align__(16) bf16_t Vs[2][64 * 64];    // [d][t_rel]
  __shared__ __align__(16) bf16_t Pt[4][16 * 64];    // per-wave P tile

  const int tid  = threadIdx.x;
  const int lane = tid & 31, wave = tid >> 5;
  const int nc = lane & 15, half = lane >> 4;
  const int bh = blockIdx.y;
  const int q0 = blockIdx.x * 64;
  const int qw = q0 + wave * 16;

  const bf16_t* qbase = qb + (size_t)bh * TT * DD;
  const bf16_t* kbase = kb + (size_t)bh * TT * DD;
  const bf16_t* vbase = vTb + (size_t)bh * DD * TT;

  // cooperative copy mapping: thread -> (row 0..63, 64B half-row)
  const int crow = tid >> 1;
  const int ccol = (tid & 1) * 32;  // elements

  auto stage_kv = [&](int buf, int kcb) {
    const bf16_t* gk = kbase + (size_t)(kcb + crow) * DD + ccol;
    const bf16_t* gv = vbase + (size_t)crow * TT + kcb + ccol;
#pragma unroll
    for (int o = 0; o < 32; o += 8) {
      async_copy16(gk + o, &Ks[buf][crow * 64 + ccol + o]);
      async_copy16(gv + o, &Vs[buf][crow * 64 + ccol + o]);
    }
  };

  v16bf a0 = load_a_frag(qbase + (size_t)qw * DD, DD, lane);
  v16bf a1 = load_a_frag(qbase + (size_t)qw * DD + 32, DD, lane);

  float mrow[8], lsum[8];
  v8f oacc[4];
#pragma unroll
  for (int r = 0; r < 8; ++r) { mrow[r] = -3.0e38f; lsum[r] = 0.0f; }
#pragma unroll
  for (int c = 0; c < 4; ++c) {
    v8f z = {0.f, 0.f, 0.f, 0.f, 0.f, 0.f, 0.f, 0.f};
    oacc[c] = z;
  }

  const int nkt = (q0 >> 6) + 1;  // causal: k tiles 0 .. q0/64
  stage_kv(0, 0);
  for (int jt = 0; jt < nkt; ++jt) {
    const int kcb = jt * 64;
    const int cur = jt & 1;
    if (jt + 1 < nkt) {
      stage_kv(cur ^ 1, (jt + 1) * 64);  // prefetch next k/v tile
      async_wait<8>();                   // current tile done
    } else {
      async_wait<0>();
    }
    __syncthreads();

    // S = q @ k^T for a 16x64 strip: batch fragment loads, then WMMA chain
    v16bf kb0[4], kb1[4];
#pragma unroll
    for (int c = 0; c < 4; ++c) {
      kb0[c] = load_b_frag(&Ks[cur][(c * 16) * 64], 64, lane);
      kb1[c] = load_b_frag(&Ks[cur][(c * 16) * 64 + 32], 64, lane);
    }
    v8f s[4];
#pragma unroll
    for (int c = 0; c < 4; ++c) {
      v8f z = {0.f, 0.f, 0.f, 0.f, 0.f, 0.f, 0.f, 0.f};
      s[c] = wmma_bf16(a0, kb0[c], z);
      s[c] = wmma_bf16(a1, kb1[c], s[c]);
    }

    // scale + causal mask + online softmax (row lives across 16 lanes)
    float pr[4][8];
#pragma unroll
    for (int r = 0; r < 8; ++r) {
      const int qrow = qw + r + 8 * half;
      float rmax = -3.0e38f;
#pragma unroll
      for (int c = 0; c < 4; ++c) {
        int kcol = kcb + c * 16 + nc;
        float v = s[c][r] * 0.125f + (kcol > qrow ? NEGBIG : 0.0f);
        pr[c][r] = v;
        rmax = fmaxf(rmax, v);
      }
#pragma unroll
      for (int off = 1; off < 16; off <<= 1)
        rmax = fmaxf(rmax, __shfl_xor(rmax, off, 32));
      float mnew = fmaxf(mrow[r], rmax);
      float esc = __expf(mrow[r] - mnew);
      mrow[r] = mnew;
      lsum[r] *= esc;
#pragma unroll
      for (int c = 0; c < 4; ++c) oacc[c][r] *= esc;
      float psum = 0.0f;
#pragma unroll
      for (int c = 0; c < 4; ++c) {
        float p = __expf(pr[c][r] - mnew);
        pr[c][r] = p;
        psum += p;
      }
#pragma unroll
      for (int off = 1; off < 16; off <<= 1)
        psum += __shfl_xor(psum, off, 32);
      lsum[r] += psum;
    }

    // re-shape P: C-layout -> LDS row-major -> A-layout fragments
#pragma unroll
    for (int r = 0; r < 8; ++r)
#pragma unroll
      for (int c = 0; c < 4; ++c)
        Pt[wave][(r + 8 * half) * 64 + c * 16 + nc] = (bf16_t)pr[c][r];
    __syncthreads();

    v16bf pa0 = load_a_frag(&Pt[wave][0], 64, lane);
    v16bf pa1 = load_a_frag(&Pt[wave][32], 64, lane);

    // O += P @ V: batch fragment loads, then WMMA chain
    v16bf vb0[4], vb1[4];
#pragma unroll
    for (int c2 = 0; c2 < 4; ++c2) {
      vb0[c2] = load_b_frag(&Vs[cur][(c2 * 16) * 64], 64, lane);
      vb1[c2] = load_b_frag(&Vs[cur][(c2 * 16) * 64 + 32], 64, lane);
    }
#pragma unroll
    for (int c2 = 0; c2 < 4; ++c2) {
      oacc[c2] = wmma_bf16(pa0, vb0[c2], oacc[c2]);
      oacc[c2] = wmma_bf16(pa1, vb1[c2], oacc[c2]);
    }
    __syncthreads();  // Ks/Vs/Pt fully consumed before next stage
  }

  const int b = bh >> 4, h = bh & (HH - 1);
#pragma unroll
  for (int r = 0; r < 8; ++r) {
    int t = qw + r + 8 * half;
    float inv = 1.0f / lsum[r];
#pragma unroll
    for (int c2 = 0; c2 < 4; ++c2)
      ob[((size_t)(b * TT + t)) * CC + h * DD + c2 * 16 + nc] =
          (bf16_t)(oacc[c2][r] * inv);
  }
}

// ---------- host launch ----------

extern "C" void kernel_launch(void* const* d_in, const int* in_sizes, int n_in,
                              void* d_out, int out_size, void* d_ws, size_t ws_size,
                              hipStream_t stream) {
  const float* x      = (const float*)d_in[0];
  // d_in[1] = mask (applied analytically)
  const float* W_attn = (const float*)d_in[2];
  const float* b_attn = (const float*)d_in[3];
  const float* W_proj = (const float*)d_in[4];
  const float* b_proj = (const float*)d_in[5];

  float* out  = (float*)d_out;                              // [B,T,C]
  float* kout = (float*)d_out + (size_t)BB * TT * CC;       // [B,H,T,D]
  float* vout = (float*)d_out + 2 * (size_t)BB * TT * CC;   // [B,H,T,D]

  char* ws = (char*)d_ws;
  const size_t NX = (size_t)BB * TT * CC;  // 8388608
  bf16_t* xb     = (bf16_t*)(ws);                          // 16 MB
  bf16_t* wqkvT  = (bf16_t*)(ws + 16777216);               // 6 MB  [3072x1024]
  bf16_t* wprojT = (bf16_t*)(ws + 23068672);               // 2 MB  [1024x1024]
  bf16_t* qb     = (bf16_t*)(ws + 25165824);               // 16 MB [B,H,T,D]
  bf16_t* kb     = (bf16_t*)(ws + 41943040);               // 16 MB [B,H,T,D]
  bf16_t* vTb    = (bf16_t*)(ws + 58720256);               // 16 MB [B,H,D,T]
  bf16_t* attnb  = (bf16_t*)(ws + 75497472);               // 16 MB [B,T,C]
  (void)ws_size; (void)in_sizes; (void)n_in; (void)out_size;

  // 1) prep: bf16 convert + weight transposes
  k_to_bf16<<<(int)((NX + 255) / 256), 256, 0, stream>>>(x, xb, (int)NX);
  k_transpose_bf16<<<(CC * 3 * CC + 255) / 256, 256, 0, stream>>>(W_attn, wqkvT, CC, 3 * CC);
  k_transpose_bf16<<<(CC * CC + 255) / 256, 256, 0, stream>>>(W_proj, wprojT, CC, CC);

  // 2) QKV GEMM (+ scatter q/k/v, write k/v f32 outputs)
  k_gemm<true><<<dim3(3 * CC / 128, BB * TT / 128), 256, 0, stream>>>(
      xb, wqkvT, b_attn, nullptr, qb, kout, kb, vout, vTb);

  // 3) flash attention per (b,h) with 64-row q tiles
  k_attn<<<dim3(TT / 64, BB * HH), 128, 0, stream>>>(qb, kb, vTb, attnb);

  // 4) output projection
  k_gemm<false><<<dim3(CC / 128, BB * TT / 128), 256, 0, stream>>>(
      attnb, wprojT, b_proj, out, nullptr, nullptr, nullptr, nullptr, nullptr);
}
